// LSTM_55997783605578
// MI455X (gfx1250) — matrix operand order repaired and checked by reading, not compiled, not run on hardware
//
#include <hip/hip_runtime.h>
#include <hip/hip_bf16.h>
#include <cstdint>

#define T_SEQ 2048
#define VOCAB 32000
#define EMB   512
#define HID   1024
#define G4    4096   // 4*HID

typedef __bf16 bf16;
typedef __bf16 bfx2  __attribute__((ext_vector_type(2)));
typedef __bf16 bfx8  __attribute__((ext_vector_type(8)));
typedef __bf16 bfx16 __attribute__((ext_vector_type(16)));
typedef float  fx8   __attribute__((ext_vector_type(8)));

// ---------------------------------------------------------------------------
// Async global -> LDS copy helpers (CDNA5 GLOBAL_LOAD_ASYNC_TO_LDS_B128)
// Builtin signature (from hipcc diagnostic): param0 = int4-vector pointer in
// global (__device__/AS1) address space; param1 = LDS (AS3) side.
// ---------------------------------------------------------------------------
#if __has_builtin(__builtin_amdgcn_global_load_async_to_lds_b128)
#define HAS_ASYNC_B128 1
#else
#define HAS_ASYNC_B128 0
#endif

typedef int v4i_gcc __attribute__((vector_size(4 * sizeof(int))));
typedef __attribute__((address_space(1))) v4i_gcc* gptr_b128;
typedef __attribute__((address_space(3))) v4i_gcc* lptr_b128;

__device__ __forceinline__ void async_copy_b128(const void* gsrc, void* ldst) {
#if HAS_ASYNC_B128
  __builtin_amdgcn_global_load_async_to_lds_b128((gptr_b128)gsrc,
                                                 (lptr_b128)ldst, 0, 0);
#else
  unsigned lofs =
      (unsigned)(unsigned long long)(__attribute__((address_space(3))) char*)ldst;
  unsigned long long ga = (unsigned long long)gsrc;
  asm volatile("global_load_async_to_lds_b128 %0, %1, off" ::"v"(lofs), "v"(ga)
               : "memory");
#endif
}

__device__ __forceinline__ void wait_async_lds() {
#if __has_builtin(__builtin_amdgcn_s_wait_asynccnt)
  __builtin_amdgcn_s_wait_asynccnt(0);
#else
  asm volatile("s_wait_asynccnt 0x0" ::: "memory");
#endif
}

// ---------------------------------------------------------------------------
// Small elementwise helpers
// ---------------------------------------------------------------------------
__global__ void to_bf16_kernel(const float* __restrict__ src,
                               bf16* __restrict__ dst, size_t n) {
  size_t i = (size_t)blockIdx.x * blockDim.x + threadIdx.x;
  size_t stride = (size_t)gridDim.x * blockDim.x;
  for (; i < n; i += stride) dst[i] = (bf16)src[i];
}

__global__ void bias_sum_kernel(const float* __restrict__ a,
                                const float* __restrict__ b,
                                float* __restrict__ out, int n) {
  int i = blockIdx.x * blockDim.x + threadIdx.x;
  if (i < n) out[i] = a[i] + b[i];
}

__global__ void embed_gather_kernel(const int* __restrict__ inputs,
                                    const float* __restrict__ emb,
                                    bf16* __restrict__ xs) {
  int t = blockIdx.x;
  int v = inputs[t];
  const float* src = emb + (size_t)v * EMB;
  bf16* dst = xs + (size_t)t * EMB;
  for (int e = threadIdx.x; e < EMB; e += blockDim.x) dst[e] = (bf16)src[e];
}

// ---------------------------------------------------------------------------
// WMMA bf16 GEMM:  C[M,N] = A[M,K] * W[N,K]^T + bias[N]   (f32 accumulate)
// Block tile 64(M) x 128(N), K step 32, 8 waves (2x4), wave tile 32x32.
// Double-buffered LDS tiles filled with async global->LDS copies.
// ---------------------------------------------------------------------------
#define BM 64
#define BN 128
#define BK 32
#define LDP 40   // LDS pitch in bf16 elements (80B = 20 banks, conflict-free)

__global__ __launch_bounds__(256) void gemm_bf16_wmma(
    const bf16* __restrict__ A, int lda,
    const bf16* __restrict__ W, int ldw,
    const float* __restrict__ bias,
    float* __restrict__ C, int M, int N, int K) {
  __shared__ bf16 a_lds[2][BM * LDP];
  __shared__ bf16 w_lds[2][BN * LDP];

  const int tid  = threadIdx.x;
  const int bn   = blockIdx.x * BN;
  const int bm   = blockIdx.y * BM;
  const int wave = tid >> 5;
  const int lane = tid & 31;
  const int wm   = (wave & 1) * 32;   // wave row offset in block tile
  const int wn   = (wave >> 1) * 32;  // wave col offset in block tile
  const int hl   = lane >> 4;         // lane half (0/1)
  const int ml   = lane & 15;

  fx8 acc[2][2];
  const fx8 zero8 = {0.f, 0.f, 0.f, 0.f, 0.f, 0.f, 0.f, 0.f};
#pragma unroll
  for (int mi = 0; mi < 2; ++mi)
#pragma unroll
    for (int ni = 0; ni < 2; ++ni) acc[mi][ni] = zero8;

  const int r_a = tid >> 2;      // 0..63
  const int kg  = tid & 3;       // 0..3  (8 bf16 each)

  // issue async fills of one K-tile into LDS buffer b
  auto load_tile = [&](int kb, int b) {
    async_copy_b128(A + (size_t)(bm + r_a) * lda + kb + kg * 8,
                    &a_lds[b][r_a * LDP + kg * 8]);
    async_copy_b128(W + (size_t)(bn + r_a) * ldw + kb + kg * 8,
                    &w_lds[b][r_a * LDP + kg * 8]);
    async_copy_b128(W + (size_t)(bn + r_a + 64) * ldw + kb + kg * 8,
                    &w_lds[b][(r_a + 64) * LDP + kg * 8]);
  };

  int buf = 0;
  load_tile(0, buf);
  wait_async_lds();
  __syncthreads();

  for (int kb = 0; kb < K; kb += BK) {
    if (kb + BK < K) load_tile(kb + BK, buf ^ 1);  // overlap with compute

    // --- gather fragments per CDNA5 ISA VGPR layout (2x b128 LDS each) ---
    bfx16 afr[2], bfr[2];
#pragma unroll
    for (int mi = 0; mi < 2; ++mi) {
      const int tr = wm + mi * 16 + ml;
#pragma unroll
      for (int v = 0; v < 8; ++v) {
        const int bk2 = ((v < 4) ? 0 : 16) + hl * 8 + (v & 3) * 2;
        bfx2 p = *(const bfx2*)&a_lds[buf][tr * LDP + bk2];
        afr[mi][2 * v] = p[0];
        afr[mi][2 * v + 1] = p[1];
      }
    }
#pragma unroll
    for (int ni = 0; ni < 2; ++ni) {
      const int tc = wn + ni * 16 + ml;
#pragma unroll
      for (int v = 0; v < 8; ++v) {
        const int bk2 = ((v < 4) ? 0 : 16) + hl * 8 + (v & 3) * 2;
        bfx2 p = *(const bfx2*)&w_lds[buf][tc * LDP + bk2];
        bfr[ni][2 * v] = p[0];
        bfr[ni][2 * v + 1] = p[1];
      }
    }
#pragma unroll
    for (int mi = 0; mi < 2; ++mi)
#pragma unroll
      for (int ni = 0; ni < 2; ++ni)
        acc[mi][ni] = __builtin_amdgcn_wmma_f32_16x16x32_bf16(
            false, afr[mi], false, bfr[ni], (short)0, acc[mi][ni], false, false);

    wait_async_lds();   // our async writes into buf^1 have landed
    __syncthreads();    // everyone done reading buf / writing buf^1
    buf ^= 1;
  }

  // --- epilogue: C/D layout  (ml -> N, lane half*8 + vgpr r -> M) ---
#pragma unroll
  for (int mi = 0; mi < 2; ++mi)
#pragma unroll
    for (int ni = 0; ni < 2; ++ni) {
      const int col = bn + wn + ni * 16 + ml;
      const float bv = bias[col];
#pragma unroll
      for (int r = 0; r < 8; ++r) {
        const int row = bm + wm + mi * 16 + hl * 8 + r;
        C[(size_t)row * N + col] = acc[mi][ni][r] + bv;
      }
    }
}

// ---------------------------------------------------------------------------
// Persistent LSTM scan. 128 blocks x 256 threads. Each block pins its 32 rows
// of w_hh (8 hidden units x 4 gates, bf16) in LDS (64KB) for all 2048 steps.
// Wave w owns hidden unit (blk*8 + w): rows {w, 8+w, 16+w, 24+w}.
// One grid-wide barrier per timestep (atomic counter + s_sleep spin).
// ---------------------------------------------------------------------------
#define NB_SCAN 128

__device__ __forceinline__ void grid_sync(int* cnt, volatile int* gen, int nb) {
  __syncthreads();
  if (threadIdx.x == 0) {
    __threadfence();
    int g = *gen;
    if (atomicAdd(cnt, 1) == nb - 1) {
      *cnt = 0;
      __threadfence();
      atomicAdd((int*)gen, 1);
    } else {
      while (*gen == g) { __builtin_amdgcn_s_sleep(1); }
    }
    __threadfence();
  }
  __syncthreads();
}

__device__ __forceinline__ float sigmoid_f(float x) {
  return 1.0f / (1.0f + __expf(-x));
}

__global__ __launch_bounds__(256) void lstm_scan_kernel(
    const float* __restrict__ pre_x,   // [T, 4H] gate-major (i,f,g,o)
    const bf16*  __restrict__ w_hh,    // [4H, H] bf16 row-major
    float* __restrict__ h_glob,        // [H] f32, live h vector
    bf16*  __restrict__ ys,            // [T, H] bf16 layer output
    float* __restrict__ h_out,         // [H] final h
    float* __restrict__ c_out,         // [H] final c
    int* __restrict__ sync) {
  const int tid  = threadIdx.x;
  const int blk  = blockIdx.x;
  const int wave = tid >> 5;
  const int lane = tid & 31;

  __shared__ bf16 w_lds[32 * HID];  // 64 KB, pinned for all timesteps

  // async pin of this block's 32 w_hh rows into LDS
  for (int i = tid; i < 32 * (HID / 8); i += 256) {
    const int lr = i >> 7;            // local row 0..31
    const int kk = (i & 127) * 8;
    const int gate = lr >> 3, uo = lr & 7;
    const bf16* src = w_hh + (size_t)(gate * HID + blk * 8 + uo) * HID + kk;
    async_copy_b128(src, &w_lds[lr * HID + kk]);
  }
  if (tid < 8) h_glob[blk * 8 + tid] = 0.0f;  // h0 = 0
  wait_async_lds();
  __syncthreads();
  grid_sync(sync, (volatile int*)(sync + 1), NB_SCAN);

  const int unit = blk * 8 + wave;    // hidden unit owned by this wave
  float c_reg = 0.0f;

  for (int t = 0; t < T_SEQ; ++t) {
    // load h into registers (coalesced across lanes), reuse for 4 gate rows
    float hreg[32];
#pragma unroll
    for (int q = 0; q < 32; ++q) hreg[q] = h_glob[lane + 32 * q];

    float gacc[4];
#pragma unroll
    for (int gate = 0; gate < 4; ++gate) {
      const int lr = gate * 8 + wave;
      float acc = 0.0f;
#pragma unroll
      for (int q = 0; q < 32; ++q)
        acc += (float)w_lds[lr * HID + lane + 32 * q] * hreg[q];
      // wave32 xor-reduction: result lands in all lanes
#pragma unroll
      for (int off = 16; off > 0; off >>= 1) acc += __shfl_xor(acc, off, 32);
      gacc[gate] = acc;
    }

    const float* px = pre_x + (size_t)t * G4;
    float ig = sigmoid_f(gacc[0] + px[unit]);
    float fg = sigmoid_f(gacc[1] + px[HID + unit]);
    float gg = tanhf(gacc[2] + px[2 * HID + unit]);
    float og = sigmoid_f(gacc[3] + px[3 * HID + unit]);
    c_reg = fg * c_reg + ig * gg;
    const float h = og * tanhf(c_reg);

    if (lane == 0) {
      h_glob[unit] = h;
      ys[(size_t)t * HID + unit] = (bf16)h;
      if (t == T_SEQ - 1) { h_out[unit] = h; c_out[unit] = c_reg; }
    }
    grid_sync(sync, (volatile int*)(sync + 1), NB_SCAN);
  }
}

// ---------------------------------------------------------------------------
// In-place row softmax over V=32000 (one block per row)
// ---------------------------------------------------------------------------
__global__ __launch_bounds__(256) void softmax_rows_kernel(float* __restrict__ x,
                                                           int n) {
  float* row = x + (size_t)blockIdx.x * n;
  const int tid = threadIdx.x;
  __shared__ float red[256];

  float m = -3.4e38f;
  for (int i = tid; i < n; i += 256) m = fmaxf(m, row[i]);
  red[tid] = m;
  __syncthreads();
  for (int s = 128; s > 0; s >>= 1) {
    if (tid < s) red[tid] = fmaxf(red[tid], red[tid + s]);
    __syncthreads();
  }
  m = red[0];
  __syncthreads();

  float sum = 0.0f;
  for (int i = tid; i < n; i += 256) {
    float e = __expf(row[i] - m);
    row[i] = e;
    sum += e;
  }
  red[tid] = sum;
  __syncthreads();
  for (int s = 128; s > 0; s >>= 1) {
    if (tid < s) red[tid] += red[tid + s];
    __syncthreads();
  }
  const float inv = 1.0f / red[0];
  for (int i = tid; i < n; i += 256) row[i] *= inv;
}

// ---------------------------------------------------------------------------
// Host launcher
// ---------------------------------------------------------------------------
extern "C" void kernel_launch(void* const* d_in, const int* in_sizes, int n_in,
                              void* d_out, int out_size, void* d_ws,
                              size_t ws_size, hipStream_t stream) {
  const int*   inputs = (const int*)d_in[0];
  const float* emb    = (const float*)d_in[1];
  const float* w_ih0  = (const float*)d_in[2];
  const float* w_hh0  = (const float*)d_in[3];
  const float* b_ih0  = (const float*)d_in[4];
  const float* b_hh0  = (const float*)d_in[5];
  const float* w_ih1  = (const float*)d_in[6];
  const float* w_hh1  = (const float*)d_in[7];
  const float* b_ih1  = (const float*)d_in[8];
  const float* b_hh1  = (const float*)d_in[9];
  const float* w_out  = (const float*)d_in[10];
  const float* b_out  = (const float*)d_in[11];
  float* out = (float*)d_out;

  // workspace carve-up (256B aligned)
  char* ws = (char*)d_ws;
  size_t off = 0;
  auto take = [&](size_t bytes) -> char* {
    char* p = ws + off;
    off += (bytes + 255) & ~(size_t)255;
    return p;
  };
  int*   sync   = (int*)take(256);
  float* h_glob = (float*)take(HID * sizeof(float));
  bf16*  xs_b   = (bf16*)take((size_t)T_SEQ * EMB * 2);
  bf16*  wih0_b = (bf16*)take((size_t)G4 * EMB * 2);
  bf16*  whh0_b = (bf16*)take((size_t)G4 * HID * 2);
  bf16*  wih1_b = (bf16*)take((size_t)G4 * HID * 2);
  bf16*  whh1_b = (bf16*)take((size_t)G4 * HID * 2);
  bf16*  wout_b = (bf16*)take((size_t)VOCAB * HID * 2);
  float* bsum0  = (float*)take(G4 * sizeof(float));
  float* bsum1  = (float*)take(G4 * sizeof(float));
  float* prex0  = (float*)take((size_t)T_SEQ * G4 * sizeof(float));
  float* prex1  = (float*)take((size_t)T_SEQ * G4 * sizeof(float));
  bf16*  ys0_b  = (bf16*)take((size_t)T_SEQ * HID * 2);
  bf16*  ys1_b  = (bf16*)take((size_t)T_SEQ * HID * 2);

  float* hout = out + (size_t)T_SEQ * VOCAB;  // h: [2,1,H]
  float* cout = hout + 2 * HID;               // c: [2,1,H]

  // ---- precision converts & bias folds ----
  to_bf16_kernel<<<4096, 256, 0, stream>>>(w_ih0, wih0_b, (size_t)G4 * EMB);
  to_bf16_kernel<<<4096, 256, 0, stream>>>(w_hh0, whh0_b, (size_t)G4 * HID);
  to_bf16_kernel<<<4096, 256, 0, stream>>>(w_ih1, wih1_b, (size_t)G4 * HID);
  to_bf16_kernel<<<4096, 256, 0, stream>>>(w_hh1, whh1_b, (size_t)G4 * HID);
  to_bf16_kernel<<<8192, 256, 0, stream>>>(w_out, wout_b, (size_t)VOCAB * HID);
  bias_sum_kernel<<<G4 / 256, 256, 0, stream>>>(b_ih0, b_hh0, bsum0, G4);
  bias_sum_kernel<<<G4 / 256, 256, 0, stream>>>(b_ih1, b_hh1, bsum1, G4);
  embed_gather_kernel<<<T_SEQ, 256, 0, stream>>>(inputs, emb, xs_b);

  // ---- layer 0: input projection (WMMA) + persistent scan ----
  gemm_bf16_wmma<<<dim3(G4 / BN, T_SEQ / BM), 256, 0, stream>>>(
      xs_b, EMB, wih0_b, EMB, bsum0, prex0, T_SEQ, G4, EMB);
  (void)hipMemsetAsync(sync, 0, 256, stream);
  lstm_scan_kernel<<<NB_SCAN, 256, 0, stream>>>(prex0, whh0_b, h_glob, ys0_b,
                                                hout, cout, sync);

  // ---- layer 1 ----
  gemm_bf16_wmma<<<dim3(G4 / BN, T_SEQ / BM), 256, 0, stream>>>(
      ys0_b, HID, wih1_b, HID, bsum1, prex1, T_SEQ, G4, HID);
  (void)hipMemsetAsync(sync, 0, 256, stream);
  lstm_scan_kernel<<<NB_SCAN, 256, 0, stream>>>(prex1, whh1_b, h_glob, ys1_b,
                                                hout + HID, cout + HID, sync);

  // ---- logits straight into d_out, softmax in place ----
  gemm_bf16_wmma<<<dim3(VOCAB / BN, T_SEQ / BM), 256, 0, stream>>>(
      ys1_b, HID, wout_b, HID, b_out, out, T_SEQ, VOCAB, HID);
  softmax_rows_kernel<<<T_SEQ, 256, 0, stream>>>(out, VOCAB);
}
